// GraphAttentionAggregator_88021059764653
// MI455X (gfx1250) — matrix operand order repaired and checked by reading, not compiled, not run on hardware
//
#include <hip/hip_runtime.h>
#include <cstdint>

typedef _Float16 f16;
typedef __attribute__((ext_vector_type(16))) _Float16 v16h;
typedef __attribute__((ext_vector_type(8)))  float    v8f;

#define NB   8      // batches
#define NN   1024   // nodes
#define DF   2048   // L*F flattened feature dim
#define NH   4      // heads

// ---------------------------------------------------------------------------
// k0: convert s_flat f32 -> f16 AND transpose to [b][d][j] so that the GEMM's
//     B-matrix fragments (per-lane column, K-pairs packed in 32b) are loaded
//     from K-contiguous LDS with aligned b32/b128 reads.
// ---------------------------------------------------------------------------
__global__ __launch_bounds__(256)
void k0_cvt_transpose(const float* __restrict__ s_seq, f16* __restrict__ s16T) {
  __shared__ f16 tile[32][33];
  const int b     = blockIdx.z;
  const int dBase = blockIdx.x * 32;
  const int jBase = blockIdx.y * 32;
  const int tx = threadIdx.x;   // 0..31
  const int ty = threadIdx.y;   // 0..7
#pragma unroll
  for (int r = 0; r < 4; ++r) {
    const int j = jBase + ty + r * 8;
    tile[ty + r * 8][tx] = (f16)s_seq[((size_t)b * NN + j) * DF + dBase + tx];
  }
  __syncthreads();
#pragma unroll
  for (int r = 0; r < 4; ++r) {
    const int d = dBase + ty + r * 8;
    s16T[((size_t)b * DF + d) * NN + jBase + tx] = tile[tx][ty + r * 8];
  }
}

// ---------------------------------------------------------------------------
// k1: wh = W @ s_last (per node), then src/dst head projections. Tiny.
// ---------------------------------------------------------------------------
__global__ __launch_bounds__(64)
void k1_wh_srcdst(const float* __restrict__ s_seq, const float* __restrict__ W,
                  const float* __restrict__ a, float* __restrict__ srcv,
                  float* __restrict__ dstv) {
  __shared__ float srow[64];
  __shared__ float wrow[64];
  const int bn = blockIdx.x;          // b*NN + n
  const int t  = threadIdx.x;         // 0..63
  srow[t] = s_seq[((size_t)bn * 32 + 31) * 64 + t];   // last timestep
  __syncthreads();
  float acc = 0.f;
#pragma unroll 8
  for (int f = 0; f < 64; ++f) acc = fmaf(W[t * 64 + f], srow[f], acc);
  wrow[t] = acc;
  __syncthreads();
  if (t < NH) {
    float s = 0.f, d = 0.f;
#pragma unroll 8
    for (int f = 0; f < 64; ++f) {
      s = fmaf(wrow[f], a[t * 128 + f], s);
      d = fmaf(wrow[f], a[t * 128 + 64 + f], d);
    }
    srcv[(size_t)bn * NH + t] = s;
    dstv[(size_t)bn * NH + t] = d;
  }
}

// ---------------------------------------------------------------------------
// k2: per (b,i): e[j,h] = leaky(src[i,h]+dst[j,h]); mask; softmax over j per
//     head; mean over heads -> alpha[b][i][j] stored as f16 (K-contiguous A).
// ---------------------------------------------------------------------------
__global__ __launch_bounds__(256)
void k2_alpha(const float4* __restrict__ srcv, const float4* __restrict__ dstv,
              const unsigned char* __restrict__ mask, f16* __restrict__ alpha) {
  __shared__ float e[NN * NH];     // 16 KB
  __shared__ float red[256 * NH];  //  4 KB
  const int bi  = blockIdx.x;      // b*NN + i
  const int b   = bi >> 10;
  const int tid = threadIdx.x;
  const float4 s = srcv[bi];
  const unsigned char* mrow = mask + (size_t)bi * NN;

  float mx0 = -INFINITY, mx1 = -INFINITY, mx2 = -INFINITY, mx3 = -INFINITY;
#pragma unroll
  for (int jj = 0; jj < 4; ++jj) {
    const int j = tid + jj * 256;
    const float4 d = dstv[b * NN + j];
    float e0 = s.x + d.x, e1 = s.y + d.y, e2 = s.z + d.z, e3 = s.w + d.w;
    e0 = e0 >= 0.f ? e0 : 0.2f * e0;   e1 = e1 >= 0.f ? e1 : 0.2f * e1;
    e2 = e2 >= 0.f ? e2 : 0.2f * e2;   e3 = e3 >= 0.f ? e3 : 0.2f * e3;
    if (!mrow[j]) { e0 = e1 = e2 = e3 = -INFINITY; }
    e[j * NH + 0] = e0; e[j * NH + 1] = e1; e[j * NH + 2] = e2; e[j * NH + 3] = e3;
    mx0 = fmaxf(mx0, e0); mx1 = fmaxf(mx1, e1);
    mx2 = fmaxf(mx2, e2); mx3 = fmaxf(mx3, e3);
  }
  red[tid * NH + 0] = mx0; red[tid * NH + 1] = mx1;
  red[tid * NH + 2] = mx2; red[tid * NH + 3] = mx3;
  __syncthreads();
  for (int sft = 128; sft > 0; sft >>= 1) {
    if (tid < sft) {
#pragma unroll
      for (int h = 0; h < NH; ++h)
        red[tid * NH + h] = fmaxf(red[tid * NH + h], red[(tid + sft) * NH + h]);
    }
    __syncthreads();
  }
  const float m0 = red[0], m1 = red[1], m2 = red[2], m3 = red[3];
  __syncthreads();   // allow red[] reuse

  float s0 = 0.f, s1 = 0.f, s2 = 0.f, s3 = 0.f;
#pragma unroll
  for (int jj = 0; jj < 4; ++jj) {
    const int j = tid + jj * 256;
    const float v0 = __expf(e[j * NH + 0] - m0);
    const float v1 = __expf(e[j * NH + 1] - m1);
    const float v2 = __expf(e[j * NH + 2] - m2);
    const float v3 = __expf(e[j * NH + 3] - m3);
    e[j * NH + 0] = v0; e[j * NH + 1] = v1; e[j * NH + 2] = v2; e[j * NH + 3] = v3;
    s0 += v0; s1 += v1; s2 += v2; s3 += v3;
  }
  red[tid * NH + 0] = s0; red[tid * NH + 1] = s1;
  red[tid * NH + 2] = s2; red[tid * NH + 3] = s3;
  __syncthreads();
  for (int sft = 128; sft > 0; sft >>= 1) {
    if (tid < sft) {
#pragma unroll
      for (int h = 0; h < NH; ++h)
        red[tid * NH + h] += red[(tid + sft) * NH + h];
    }
    __syncthreads();
  }
  const float i0 = 0.25f / red[0], i1 = 0.25f / red[1];
  const float i2 = 0.25f / red[2], i3 = 0.25f / red[3];
  f16* arow = alpha + (size_t)bi * NN;
#pragma unroll
  for (int jj = 0; jj < 4; ++jj) {
    const int j = tid + jj * 256;
    arow[j] = (f16)(e[j * NH + 0] * i0 + e[j * NH + 1] * i1 +
                    e[j * NH + 2] * i2 + e[j * NH + 3] * i3);
  }
}

// ---------------------------------------------------------------------------
// k3: batched GEMM z[b] = alpha[b] (1024x1024 f16) x s_flat[b] (1024x2048 f16)
//     via v_wmma_f32_16x16x32_f16 with a TRIPLE-buffered ASYNC global->LDS
//     pipeline (global_load_async_to_lds_b128 / ASYNCcnt), ONE barrier per
//     K-step, fused exact GELU. 256 threads = 8 wave32s (2x4), tile 64x128.
// ---------------------------------------------------------------------------
#define KSTEP 32
#define KITER (NN / KSTEP)
#define ASTR  40   // halves; 80B row stride keeps b128 LDS accesses aligned
#define BSTR  40

union FragAB { v16h h; uint32_t u[8]; };

// CDNA5 async global->LDS copy (16B per lane, tracked by ASYNCcnt).
__device__ __forceinline__ void async_copy_b128(uint32_t lds_byte_addr,
                                                uint64_t gaddr) {
  asm volatile("global_load_async_to_lds_b128 %0, %1, off"
               :: "v"(lds_byte_addr), "v"(gaddr) : "memory");
}
__device__ __forceinline__ void wait_async_le3() {
  asm volatile("s_wait_asynccnt 0x3" ::: "memory");
}
__device__ __forceinline__ void wait_async_0() {
  asm volatile("s_wait_asynccnt 0x0" ::: "memory");
}

__global__ __launch_bounds__(256)
void k3_gemm_gelu(const f16* __restrict__ alpha, const f16* __restrict__ s16T,
                  float* __restrict__ out) {
  __shared__ alignas(16) f16 Alds[3][64 * ASTR];    // 3 x  5 KB
  __shared__ alignas(16) f16 Blds[3][128 * BSTR];   // 3 x 10 KB
  const int b     = blockIdx.z;
  const int mBase = blockIdx.y * 64;
  const int nBase = blockIdx.x * 128;
  const int tid   = threadIdx.x;
  const int lane  = tid & 31;
  const int wave  = tid >> 5;
  const int waveM = (wave >> 2) * 32;   // 0,32
  const int waveN = (wave & 3) * 32;    // 0,32,64,96
  const int lmod  = lane & 15;
  const int lhalf = lane >> 4;          // 0/1: K-half selector per ISA layout

  const f16* gA = alpha + ((size_t)b * NN + mBase) * NN;      // row-major [i][j]
  const f16* gB = s16T  + ((size_t)b * DF + nBase) * NN;      // row-major [d][j]

  // cooperative copy indices: A: 64 rows x 32 halves; B: 128 cols x 32 halves
  const int arow = tid >> 2, acol = (tid & 3) * 8;
  const int bcol = tid >> 1, bk   = (tid & 1) * 16;

  const uint64_t ga = (uint64_t)(uintptr_t)(gA + (size_t)arow * NN + acol);
  const uint64_t gb = (uint64_t)(uintptr_t)(gB + (size_t)bcol * NN + bk);
  uint32_t aL[3], bL[3];
#pragma unroll
  for (int s = 0; s < 3; ++s) {
    aL[s] = (uint32_t)(uintptr_t)&Alds[s][arow * ASTR + acol];
    bL[s] = (uint32_t)(uintptr_t)&Blds[s][bcol * BSTR + bk];
  }

  v8f acc[2][2] = {};   // plain SSA vectors -> no union shuffling

  // prologue: stage K-tiles 0 and 1 (3 async copies per thread per tile)
  async_copy_b128(aL[0], ga);
  async_copy_b128(bL[0], gb);
  async_copy_b128(bL[0] + 16, gb + 16);
  async_copy_b128(aL[1], ga + KSTEP * 2);
  async_copy_b128(bL[1], gb + KSTEP * 2);
  async_copy_b128(bL[1] + 16, gb + KSTEP * 2 + 16);

  int buf = 0;
  for (int it = 0; it < KITER; ++it) {
    // tile `it` complete when <=1 tile (3 copies) remains in flight
    if (it + 1 < KITER) wait_async_le3(); else wait_async_0();
    __syncthreads();   // publish tile `it`; all waves done reading buf(it-1)

    if (it + 2 < KITER) {
      int nb = buf + 2; if (nb >= 3) nb -= 3;   // (it+2)%3
      const uint64_t ga2 = ga + (uint64_t)(it + 2) * (KSTEP * 2);
      const uint64_t gb2 = gb + (uint64_t)(it + 2) * (KSTEP * 2);
      async_copy_b128(aL[nb], ga2);
      async_copy_b128(bL[nb], gb2);
      async_copy_b128(bL[nb] + 16, gb2 + 16);
    }

    FragAB af[2], bf[2];
#pragma unroll
    for (int tm = 0; tm < 2; ++tm) {
      const int r = waveM + tm * 16 + lmod;
      const uint32_t* base = (const uint32_t*)(&Alds[buf][r * ASTR]);
#pragma unroll
      for (int i = 0; i < 8; ++i) {
        // ISA 16-bit A 16x32: v0..3 = K{0..7}+8*lhalf, v4..7 = K{16..23}+8*lhalf
        const int kh = ((i < 4) ? (i * 2) : (16 + (i - 4) * 2)) + lhalf * 8;
        af[tm].u[i] = base[kh >> 1];
      }
    }
#pragma unroll
    for (int tn = 0; tn < 2; ++tn) {
      const int c = waveN + tn * 16 + lmod;
      const uint32_t* base = (const uint32_t*)(&Blds[buf][c * BSTR]);
#pragma unroll
      for (int i = 0; i < 8; ++i) {
        // B 32x16: v0..3 = K{0..7}+8*lhalf, v4..7 = K{16..23}+8*lhalf (matches A)
        const int kh = ((i < 4) ? (i * 2) : (16 + (i - 4) * 2)) + lhalf * 8;
        bf[tn].u[i] = base[kh >> 1];
      }
    }
#pragma unroll
    for (int tm = 0; tm < 2; ++tm)
#pragma unroll
      for (int tn = 0; tn < 2; ++tn)
        acc[tm][tn] = __builtin_amdgcn_wmma_f32_16x16x32_f16(
            false, af[tm].h, false, bf[tn].h,
            (short)0, acc[tm][tn], false, false);

    ++buf; if (buf >= 3) buf = 0;
  }

  // epilogue: exact GELU, f32 store. D layout: VGPR r -> M=r (lanes0-15), M=r+8
#pragma unroll
  for (int tm = 0; tm < 2; ++tm) {
    const int row0 = mBase + waveM + tm * 16 + lhalf * 8;
#pragma unroll
    for (int tn = 0; tn < 2; ++tn) {
      const int col = nBase + waveN + tn * 16 + lmod;
#pragma unroll
      for (int r = 0; r < 8; ++r) {
        const int row = row0 + r;
        const float z = acc[tm][tn][r];
        const float g = 0.5f * z * (1.f + erff(z * 0.70710678118654752f));
        out[((size_t)b * NN + row) * DF + col] = g;
      }
    }
  }
}

// ---------------------------------------------------------------------------
extern "C" void kernel_launch(void* const* d_in, const int* in_sizes, int n_in,
                              void* d_out, int out_size, void* d_ws, size_t ws_size,
                              hipStream_t stream) {
  const float*         s_seq = (const float*)d_in[0];
  const unsigned char* emask = (const unsigned char*)d_in[1];  // bool mask
  const float*         W     = (const float*)d_in[2];
  const float*         a     = (const float*)d_in[3];
  float*               out   = (float*)d_out;

  char* ws = (char*)d_ws;
  f16*   s16T  = (f16*)(ws);                                   // 33,554,432 B
  f16*   alpha = (f16*)(ws + 33554432);                        // 16,777,216 B
  float* srcv  = (float*)(ws + 33554432 + 16777216);           //    131,072 B
  float* dstv  = (float*)(ws + 33554432 + 16777216 + 131072);  //    131,072 B

  k0_cvt_transpose<<<dim3(DF / 32, NN / 32, NB), dim3(32, 8), 0, stream>>>(s_seq, s16T);
  k1_wh_srcdst<<<NB * NN, 64, 0, stream>>>(s_seq, W, a, srcv, dstv);
  k2_alpha<<<NB * NN, 256, 0, stream>>>((const float4*)srcv, (const float4*)dstv,
                                        emask, alpha);
  k3_gemm_gelu<<<dim3(DF / 128, NN / 64, NB), 256, 0, stream>>>(alpha, s16T, out);
}
